// MultiheadAttention_33217277067770
// MI455X (gfx1250) — compile-verified, matrix operand order checked by reading, not run on hardware
//
#include <hip/hip_runtime.h>
#include <hip/hip_bf16.h>

// ---------------- problem constants ----------------
static constexpr int B = 2;
static constexpr int S = 2048;
static constexpr int E = 1024;
static constexpr int H = 16;
static constexpr int D = 64;          // head dim

typedef __attribute__((ext_vector_type(16))) _Float16 v16h;
typedef __attribute__((ext_vector_type(8)))  _Float16 v8h;
typedef __attribute__((ext_vector_type(8)))  float    v8f;
typedef __attribute__((ext_vector_type(4)))  unsigned int u32x4;
typedef __attribute__((ext_vector_type(8)))  int i32x8;
typedef __attribute__((ext_vector_type(4)))  int i32x4;

// Tensor Data Mover availability (this toolchain: 6-arg builtin)
#if defined(__has_builtin)
#if __has_builtin(__builtin_amdgcn_tensor_load_to_lds) && \
    __has_builtin(__builtin_amdgcn_s_wait_tensorcnt)
#define USE_TDM 1
#endif
#endif
#ifndef USE_TDM
#define USE_TDM 0
#endif

// ---------------- WMMA helpers ----------------
__device__ __forceinline__ v8f wmma16(v16h a, v16h b, v8f c) {
  return __builtin_amdgcn_wmma_f32_16x16x32_f16(
      false, a, false, b, (short)0, c, false, false);
}

// 16x32 f16 fragment load (A-layout; also B when source is N-major/K-contig).
//   lanes 0-15 : row = lane,    halves 0-7 = K 0..7,  8-15 = K 16..23
//   lanes 16-31: row = lane-16, halves 0-7 = K 8..15, 8-15 = K 24..31
__device__ __forceinline__ v16h load_frag(const _Float16* src, int ld) {
  const int lane = threadIdx.x & 31;
  const int r  = lane & 15;
  const int kb = (lane < 16) ? 0 : 8;
  const _Float16* p = src + (size_t)r * ld + kb;
  v8h lo = *reinterpret_cast<const v8h*>(p);
  v8h hi = *reinterpret_cast<const v8h*>(p + 16);
  v16h out;
#pragma unroll
  for (int i = 0; i < 8; ++i) { out[i] = lo[i]; out[i + 8] = hi[i]; }
  return out;
}

#if USE_TDM
// 2-D TDM tile load: tile (tile_d0 x tile_d1) of 2-byte elements from a
// row-major tensor (tensor_d0 wide, tensor_d1 rows, stride0 elems/row)
// into LDS at lds_addr.  D# packing per CDNA5 ISA 8.3/8.4.
__device__ __forceinline__ void tdm_load_2d(unsigned lds_addr, const void* gaddr,
                                            unsigned tensor_d0, unsigned tensor_d1,
                                            unsigned tile_d0, unsigned tile_d1,
                                            unsigned stride0) {
  const unsigned long long ga = (unsigned long long)(size_t)gaddr;
  u32x4 g0;
  g0[0] = 1u;                                          // count=1, user desc
  g0[1] = lds_addr;                                    // lds_addr [63:32]
  g0[2] = (unsigned)(ga & 0xFFFFFFFFu);                // global_addr lo
  g0[3] = (unsigned)((ga >> 32) & 0x1FFFFFFu) | 0x80000000u;  // hi + type=2
  i32x8 g1;
  g1[0] = (int)(1u << 16);                             // data_size = 2 bytes
  g1[1] = (int)((tensor_d0 & 0xFFFFu) << 16);          // dim0 lo16 @ bit48
  g1[2] = (int)((tensor_d0 >> 16) | ((tensor_d1 & 0xFFFFu) << 16));
  g1[3] = (int)((tensor_d1 >> 16) | (tile_d0 << 16));  // tile_dim0 @ bit112
  g1[4] = (int)(tile_d1 & 0xFFFFu);                    // tile_dim1, tile_dim2=0
  g1[5] = (int)stride0;                                // dim0_stride lo32
  g1[6] = 0;
  g1[7] = 0;
  const i32x4 z4 = {0, 0, 0, 0};
  const i32x8 z8 = {0, 0, 0, 0, 0, 0, 0, 0};
  __builtin_amdgcn_tensor_load_to_lds(g0, g1, z4, z4, z8, 0);
}
#endif

// ---------------- shared GEMM mainloop ----------------
// C(128x64) += A(128xE) . B^T rows(64xE)   A,B K-contiguous, K = E
// block = 256 threads (8 waves, 16 rows each); lA: 2*128*64, lB: 2*64*64 halves
__device__ __forceinline__ void gemm_mainloop_128x64(
    const _Float16* __restrict__ gA, unsigned rowsA,
    const _Float16* __restrict__ gB, unsigned rowsB,
    _Float16* lA, _Float16* lB, v8f acc[4]) {
  const int tid  = threadIdx.x;
  const int wave = tid >> 5;
  const int NK   = E / 64;     // 16 k-tiles of 64
#if USE_TDM
  if (tid < 32) {
    tdm_load_2d((unsigned)(size_t)lA, gA, E, rowsA, 64, 128, E);
    tdm_load_2d((unsigned)(size_t)lB, gB, E, rowsB, 64, 64, E);
    __builtin_amdgcn_s_wait_tensorcnt(0);
  }
  __syncthreads();
  for (int t = 0; t < NK; ++t) {
    const int cur = t & 1, nxt = cur ^ 1;
    if ((t + 1) < NK && tid < 32) {
      tdm_load_2d((unsigned)(size_t)(lA + nxt * (128 * 64)),
                  gA + (t + 1) * 64, E, rowsA, 64, 128, E);
      tdm_load_2d((unsigned)(size_t)(lB + nxt * (64 * 64)),
                  gB + (t + 1) * 64, E, rowsB, 64, 64, E);
    }
    const _Float16* At = lA + cur * (128 * 64) + wave * (16 * 64);
    const _Float16* Bt = lB + cur * (64 * 64);
    // hoist all fragment loads, then issue 8 back-to-back WMMAs
    const v16h a0 = load_frag(At, 64);
    const v16h a1 = load_frag(At + 32, 64);
    v16h bf[8];
#pragma unroll
    for (int n = 0; n < 4; ++n) {
      bf[n]     = load_frag(Bt + (n * 16) * 64, 64);
      bf[n + 4] = load_frag(Bt + (n * 16) * 64 + 32, 64);
    }
#pragma unroll
    for (int n = 0; n < 4; ++n) acc[n] = wmma16(a0, bf[n], acc[n]);
#pragma unroll
    for (int n = 0; n < 4; ++n) acc[n] = wmma16(a1, bf[n + 4], acc[n]);
    if ((t + 1) < NK) {
      if (tid < 32) __builtin_amdgcn_s_wait_tensorcnt(0);
      __syncthreads();
    }
  }
#else
  // fallback: cooperative single-buffer staging
  for (int t = 0; t < NK; ++t) {
    __syncthreads();
    for (int i = tid; i < 1024; i += 256) {   // A tile: 128 rows x 8 v8h
      const int row = i >> 3, c = i & 7;
      reinterpret_cast<v8h*>(lA)[i] =
          *reinterpret_cast<const v8h*>(gA + (size_t)row * E + t * 64 + c * 8);
    }
    for (int i = tid; i < 512; i += 256) {    // B tile: 64 rows x 8 v8h
      const int row = i >> 3, c = i & 7;
      reinterpret_cast<v8h*>(lB)[i] =
          *reinterpret_cast<const v8h*>(gB + (size_t)row * E + t * 64 + c * 8);
    }
    __syncthreads();
    const _Float16* At = lA + wave * (16 * 64);
    const v16h a0 = load_frag(At, 64);
    const v16h a1 = load_frag(At + 32, 64);
    v16h bf[8];
#pragma unroll
    for (int n = 0; n < 4; ++n) {
      bf[n]     = load_frag(lB + (n * 16) * 64, 64);
      bf[n + 4] = load_frag(lB + (n * 16) * 64 + 32, 64);
    }
#pragma unroll
    for (int n = 0; n < 4; ++n) acc[n] = wmma16(a0, bf[n], acc[n]);
#pragma unroll
    for (int n = 0; n < 4; ++n) acc[n] = wmma16(a1, bf[n + 4], acc[n]);
  }
  (void)rowsA; (void)rowsB;
#endif
}

// ---------------- LayerNorm -> f16 ----------------
__global__ void __launch_bounds__(256) ln_kernel(const float* __restrict__ x,
                                                 const float* __restrict__ g,
                                                 const float* __restrict__ be,
                                                 _Float16* __restrict__ nx) {
  __shared__ float s_sum[256];
  __shared__ float s_sq[256];
  const int t = blockIdx.x;
  const float* xr = x + (size_t)t * E;
  float v[4];
  float sum = 0.f, sq = 0.f;
#pragma unroll
  for (int i = 0; i < 4; ++i) {
    v[i] = xr[threadIdx.x + i * 256];
    sum += v[i];
    sq  += v[i] * v[i];
  }
  s_sum[threadIdx.x] = sum;
  s_sq[threadIdx.x]  = sq;
  __syncthreads();
  for (int off = 128; off > 0; off >>= 1) {
    if ((int)threadIdx.x < off) {
      s_sum[threadIdx.x] += s_sum[threadIdx.x + off];
      s_sq[threadIdx.x]  += s_sq[threadIdx.x + off];
    }
    __syncthreads();
  }
  const float mu   = s_sum[0] * (1.0f / E);
  const float var  = s_sq[0] * (1.0f / E) - mu * mu;
  const float rstd = rsqrtf(var + 1e-5f);
#pragma unroll
  for (int i = 0; i < 4; ++i) {
    const int c = threadIdx.x + i * 256;
    nx[(size_t)t * E + c] = (_Float16)((v[i] - mu) * rstd * g[c] + be[c]);
  }
}

// ---------------- weight converts ----------------
// [H,E,D] f32 -> [H,D,E] f16 (K-contiguous)
__global__ void __launch_bounds__(256) wT_convert(const float* __restrict__ W,
                                                  _Float16* __restrict__ Wt) {
  const size_t i = (size_t)blockIdx.x * blockDim.x + threadIdx.x;
  if (i >= (size_t)H * D * E) return;
  const int h   = (int)(i / ((size_t)D * E));
  const int rem = (int)(i % ((size_t)D * E));
  const int d   = rem / E;
  const int e   = rem % E;
  Wt[i] = (_Float16)W[((size_t)h * E + e) * D + d];
}

__global__ void __launch_bounds__(256) h_convert(const float* __restrict__ W,
                                                 _Float16* __restrict__ o,
                                                 int n) {
  const size_t i = (size_t)blockIdx.x * blockDim.x + threadIdx.x;
  if (i < (size_t)n) o[i] = (_Float16)W[i];
}

// ---------------- fused QKV projection GEMM ----------------
// Wt = [3*H*D, E] (q rows, then k rows, then v rows), K-contiguous.
// grid: (B*S/128, 3*E/64)
__global__ void __launch_bounds__(256) qkv_gemm(const _Float16* __restrict__ nx,
                                                const _Float16* __restrict__ Wt,
                                                const float* __restrict__ bq,
                                                const float* __restrict__ bk,
                                                const float* __restrict__ bv,
                                                _Float16* __restrict__ Qb,
                                                _Float16* __restrict__ Kb,
                                                _Float16* __restrict__ Vb) {
  __shared__ __align__(16) _Float16 lA[2 * 128 * 64];
  __shared__ __align__(16) _Float16 lB[2 * 64 * 64];
  const int m0 = blockIdx.x * 128;
  const int n0 = blockIdx.y * 64;
  const int wave = threadIdx.x >> 5, lane = threadIdx.x & 31;

  v8f acc[4] = {};
  gemm_mainloop_128x64(nx + (size_t)m0 * E, (unsigned)(B * S),
                       Wt + (size_t)n0 * E, (unsigned)(3 * H * D),
                       lA, lB, acc);

  const int which = n0 >> 10;                // 0:q 1:k 2:v  (uniform)
  const int h     = (n0 & 1023) >> 6;
  const float scale = (which == 0) ? 0.125f : 1.0f;   // 1/sqrt(D) folded in Q
  const float* bias = (which == 0) ? bq : (which == 1) ? bk : bv;
  _Float16*    outp = (which == 0) ? Qb : (which == 1) ? Kb : Vb;

  const int col0 = lane & 15, rowoff = (lane >> 4) * 8;
#pragma unroll
  for (int n = 0; n < 4; ++n) {
    const int d  = n * 16 + col0;
    const float bvv = bias[h * D + d];
#pragma unroll
    for (int r = 0; r < 8; ++r) {
      const int m = m0 + wave * 16 + rowoff + r;
      const int b = m >> 11, s = m & (S - 1);
      outp[(((size_t)b * H + h) * S + s) * D + d] =
          (_Float16)((acc[n][r] + bvv) * scale);
    }
  }
}

// ---------------- Flash attention ----------------
// Q pre-scaled.  grid: (S/128, H, B), block 256 (8 waves).
// out[b, s, h*64+d] (concat heads, f16).  K/V tile staging is software-
// pipelined: next tile's global loads are issued before the current tile's
// WMMA/softmax work.
__global__ void __launch_bounds__(256) attn_kernel(const _Float16* __restrict__ Qm,
                                                   const _Float16* __restrict__ Km,
                                                   const _Float16* __restrict__ Vm,
                                                   const int* __restrict__ mask,
                                                   _Float16* __restrict__ out) {
  __shared__ __align__(16) _Float16 lK[64 * 64];       // [key][d]
  __shared__ __align__(16) _Float16 lVt[64 * 64];      // [d][key]
  __shared__ __align__(16) _Float16 lP[8 * 16 * 64];   // per-wave [row][key]

  const int qt   = blockIdx.x;
  const int h    = blockIdx.y;
  const int b    = blockIdx.z;
  const int tid  = threadIdx.x;
  const int wave = tid >> 5;
  const int lane = tid & 31;
  const int col0   = lane & 15;
  const int rowoff = (lane >> 4) * 8;

  const size_t bh = ((size_t)b * H + h) * S;
  const int m0 = qt * 128 + wave * 16;

  const _Float16* qbase = Qm + (bh + m0) * D;
  const v16h qa0 = load_frag(qbase, D);
  const v16h qa1 = load_frag(qbase + 32, D);

  const _Float16* gKb = Km + bh * D;
  const _Float16* gVb = Vm + bh * D;

  v8f oacc[4] = {};
  float rmax[8], rsum[8];
#pragma unroll
  for (int r = 0; r < 8; ++r) { rmax[r] = -1e30f; rsum[r] = 0.f; }

  // register prefetch of a 64-key tile: 16 halves of K and V per thread
  v8h kreg0, kreg1, vreg0, vreg1;
  auto prefetch = [&](int t0) {
    const v8h* gk = reinterpret_cast<const v8h*>(gKb + (size_t)t0 * D);
    kreg0 = gk[2 * tid];
    kreg1 = gk[2 * tid + 1];
    const v8h* gv = reinterpret_cast<const v8h*>(gVb + (size_t)t0 * D);
    vreg0 = gv[2 * tid];
    vreg1 = gv[2 * tid + 1];
  };
  prefetch(0);

  for (int t0 = 0; t0 < S; t0 += 64) {
    // ---- commit prefetched tile to LDS (K as-is, V transposed) ----
    reinterpret_cast<v8h*>(lK)[2 * tid]     = kreg0;
    reinterpret_cast<v8h*>(lK)[2 * tid + 1] = kreg1;
    {
      const int key = tid >> 2, d0 = (tid & 3) * 16;
#pragma unroll
      for (int j = 0; j < 8; ++j) {
        lVt[(d0 + j) * 64 + key]     = vreg0[j];
        lVt[(d0 + 8 + j) * 64 + key] = vreg1[j];
      }
    }
    __syncthreads();
    if (t0 + 64 < S) prefetch(t0 + 64);   // overlaps the compute below

    // ---- scores: Q . K^T  (hoisted frags, 8 back-to-back WMMAs) ----
    v16h kb[8];
#pragma unroll
    for (int n = 0; n < 4; ++n) {
      kb[n]     = load_frag(lK + (n * 16) * 64, 64);
      kb[n + 4] = load_frag(lK + (n * 16) * 64 + 32, 64);
    }
    v8f sacc[4] = {};
#pragma unroll
    for (int n = 0; n < 4; ++n) sacc[n] = wmma16(qa0, kb[n], sacc[n]);
#pragma unroll
    for (int n = 0; n < 4; ++n) sacc[n] = wmma16(qa1, kb[n + 4], sacc[n]);

    // ---- key mask ----
#pragma unroll
    for (int n = 0; n < 4; ++n) {
      const int key = t0 + n * 16 + col0;
      if (mask[b * S + key] == 0) {
#pragma unroll
        for (int r = 0; r < 8; ++r) sacc[n][r] = -1e9f;
      }
    }
    // ---- online softmax ----
    float corr[8];
#pragma unroll
    for (int r = 0; r < 8; ++r) {
      float tm = fmaxf(fmaxf(sacc[0][r], sacc[1][r]),
                       fmaxf(sacc[2][r], sacc[3][r]));
      for (int off = 8; off >= 1; off >>= 1)
        tm = fmaxf(tm, __shfl_xor(tm, off, 32));
      const float nm = fmaxf(rmax[r], tm);
      corr[r] = __expf(rmax[r] - nm);
      rmax[r] = nm;
    }
    float tsum[8] = {};
#pragma unroll
    for (int n = 0; n < 4; ++n) {
#pragma unroll
      for (int r = 0; r < 8; ++r) {
        const float p = __expf(sacc[n][r] - rmax[r]);
        sacc[n][r] = p;
        tsum[r] += p;
      }
    }
#pragma unroll
    for (int r = 0; r < 8; ++r) {
      float ts = tsum[r];
      for (int off = 8; off >= 1; off >>= 1) ts += __shfl_xor(ts, off, 32);
      rsum[r] = rsum[r] * corr[r] + ts;
#pragma unroll
      for (int n = 0; n < 4; ++n) oacc[n][r] *= corr[r];
    }
    // ---- P: C-layout -> LDS -> A-layout ----
    _Float16* pw = lP + wave * (16 * 64);
#pragma unroll
    for (int n = 0; n < 4; ++n) {
#pragma unroll
      for (int r = 0; r < 8; ++r)
        pw[(rowoff + r) * 64 + n * 16 + col0] = (_Float16)sacc[n][r];
    }
    __syncthreads();

    // ---- O += P . V  (hoisted frags) ----
    const v16h pa0 = load_frag(pw, 64);
    const v16h pa1 = load_frag(pw + 32, 64);
    v16h vb[8];
#pragma unroll
    for (int n = 0; n < 4; ++n) {
      vb[n]     = load_frag(lVt + (n * 16) * 64, 64);
      vb[n + 4] = load_frag(lVt + (n * 16) * 64 + 32, 64);
    }
#pragma unroll
    for (int n = 0; n < 4; ++n) oacc[n] = wmma16(pa0, vb[n], oacc[n]);
#pragma unroll
    for (int n = 0; n < 4; ++n) oacc[n] = wmma16(pa1, vb[n + 4], oacc[n]);
    __syncthreads();
  }

  // ---- normalize + store ----
  _Float16* obase = out + ((size_t)b * S + m0) * E + h * D;
#pragma unroll
  for (int r = 0; r < 8; ++r) {
    const float inv = 1.0f / rsum[r];
#pragma unroll
    for (int n = 0; n < 4; ++n)
      obase[(size_t)(rowoff + r) * E + n * 16 + col0] =
          (_Float16)(oacc[n][r] * inv);
  }
}

// ---------------- output projection + residual ----------------
// y[m,e] = A[m,:] . Wp[e,:] + bp[e] + x[m,e]
// grid: (M/128, E/64)
__global__ void __launch_bounds__(256) proj_kernel(const _Float16* __restrict__ A,
                                                   const _Float16* __restrict__ W,
                                                   const float* __restrict__ bp,
                                                   const float* __restrict__ x,
                                                   float* __restrict__ y) {
  __shared__ __align__(16) _Float16 lA[2 * 128 * 64];
  __shared__ __align__(16) _Float16 lB[2 * 64 * 64];
  const int m0 = blockIdx.x * 128;
  const int n0 = blockIdx.y * 64;
  const int wave = threadIdx.x >> 5, lane = threadIdx.x & 31;

  v8f acc[4] = {};
  gemm_mainloop_128x64(A + (size_t)m0 * E, (unsigned)(B * S),
                       W + (size_t)n0 * E, (unsigned)E,
                       lA, lB, acc);

  const int col0 = lane & 15, rowoff = (lane >> 4) * 8;
#pragma unroll
  for (int n = 0; n < 4; ++n) {
    const int c = n0 + n * 16 + col0;
    const float bias = bp[c];
#pragma unroll
    for (int r = 0; r < 8; ++r) {
      const size_t idx = (size_t)(m0 + wave * 16 + rowoff + r) * E + c;
      y[idx] = acc[n][r] + bias + x[idx];
    }
  }
}

// ---------------- launcher ----------------
extern "C" void kernel_launch(void* const* d_in, const int* in_sizes, int n_in,
                              void* d_out, int out_size, void* d_ws, size_t ws_size,
                              hipStream_t stream) {
  const float* x    = (const float*)d_in[0];
  const int*   amsk = (const int*)  d_in[1];
  const float* ln_g = (const float*)d_in[2];
  const float* ln_b = (const float*)d_in[3];
  const float* Wq   = (const float*)d_in[4];
  const float* bq   = (const float*)d_in[5];
  const float* Wk   = (const float*)d_in[6];
  const float* bk   = (const float*)d_in[7];
  const float* Wv   = (const float*)d_in[8];
  const float* bv   = (const float*)d_in[9];
  const float* Wp   = (const float*)d_in[10];
  const float* bp   = (const float*)d_in[11];
  float* out = (float*)d_out;

  char* ws = (char*)d_ws;
  _Float16* nx     = (_Float16*)(ws + 0);            //  8 MB [B*S, E]
  _Float16* WqkvT  = (_Float16*)(ws + (8u  << 20));  //  6 MB [3*H*D, E]
  _Float16* WpH    = (_Float16*)(ws + (14u << 20));  //  2 MB [E, E]
  _Float16* Qb     = (_Float16*)(ws + (16u << 20));  //  8 MB [B,H,S,D]
  _Float16* Kb     = (_Float16*)(ws + (24u << 20));
  _Float16* Vb     = (_Float16*)(ws + (32u << 20));
  _Float16* AO     = (_Float16*)(ws + (40u << 20));  //  8 MB [B*S, E]

  // 1) LayerNorm -> f16
  ln_kernel<<<B * S, 256, 0, stream>>>(x, ln_g, ln_b, nx);

  // 2) weight converts (Q/K/V stacked into one [3*H*D, E] matrix)
  const int nW = H * D * E;
  wT_convert<<<(nW + 255) / 256, 256, 0, stream>>>(Wq, WqkvT);
  wT_convert<<<(nW + 255) / 256, 256, 0, stream>>>(Wk, WqkvT + (size_t)nW);
  wT_convert<<<(nW + 255) / 256, 256, 0, stream>>>(Wv, WqkvT + 2 * (size_t)nW);
  h_convert<<<(E * E + 255) / 256, 256, 0, stream>>>(Wp, WpH, E * E);

  // 3) fused QKV projection
  dim3 gqkv((B * S) / 128, (3 * E) / 64);
  qkv_gemm<<<gqkv, 256, 0, stream>>>(nx, WqkvT, bq, bk, bv, Qb, Kb, Vb);

  // 4) flash attention
  dim3 ga(S / 128, H, B);
  attn_kernel<<<ga, 256, 0, stream>>>(Qb, Kb, Vb, amsk, AO);

  // 5) output projection + bias + residual
  dim3 gp((B * S) / 128, E / 64);
  proj_kernel<<<gp, 256, 0, stream>>>(AO, WpH, bp, x, out);

  (void)in_sizes; (void)n_in; (void)out_size; (void)ws_size;
}
